// _ReservoirExtractor_57303453663386
// MI455X (gfx1250) — matrix operand order repaired and checked by reading, not compiled, not run on hardware
//
#include <hip/hip_runtime.h>
#include <math.h>

#define VOCAB  50257
#define EMBED  512
#define RES    2048
#define BATCH  64
#define SEQ    512
#define LN_EPS 1e-5f

// ---------------------------------------------------------------------------
// WMMA fragment types (CDNA5 wave32, V_WMMA_F32_16X16X32_BF16)
// ---------------------------------------------------------------------------
typedef __attribute__((ext_vector_type(16))) __bf16        v16bf;
typedef __attribute__((ext_vector_type(8)))  float         v8f;
typedef __attribute__((ext_vector_type(8)))  unsigned short us8;

union BFrag { v16bf v; us8 h[2]; };

static __device__ __forceinline__ unsigned short f2bf(float f) {
  // round-to-nearest-even f32 -> bf16
  unsigned u = __float_as_uint(f);
  unsigned r = (u + 0x7FFFu + ((u >> 16) & 1u)) >> 16;
  return (unsigned short)r;
}

static __device__ __forceinline__ float fast_tanh(float v) {
  // tanh(v) = 1 - 2/(exp(2v)+1): v_exp_f32 + v_rcp_f32, saturates cleanly
  // (exp->inf gives 1.0, exp->0 gives -1.0; no NaN at the tails)
  const float e = __expf(2.0f * v);
  return 1.0f - 2.0f * __builtin_amdgcn_rcpf(e + 1.0f);
}

// ---------------------------------------------------------------------------
// K-panel accumulate: two interleaved WMMA chains over [0, klen)
// aRow/bRow already point at the row base (row*stride included).
// A layout: lane m+16h holds K in {16h..16h+15} of row m
// B layout: lane n+16h holds K in {16h..16h+15} of column n
// ---------------------------------------------------------------------------
static __device__ __forceinline__ void panel_accum(
    v8f& acc0, v8f& acc1,
    const unsigned short* __restrict__ aRow,
    const unsigned short* __restrict__ bRow,
    int klen, int h) {
#pragma unroll 4
  for (int k0 = 0; k0 < klen; k0 += 64) {
    BFrag a0, b0, a1, b1;
    a0.h[0] = *(const us8*)(aRow + k0 + 16 * h);
    a0.h[1] = *(const us8*)(aRow + k0 + 16 * h + 8);
    b0.h[0] = *(const us8*)(bRow + k0 + 16 * h);
    b0.h[1] = *(const us8*)(bRow + k0 + 16 * h + 8);
    a1.h[0] = *(const us8*)(aRow + k0 + 32 + 16 * h);
    a1.h[1] = *(const us8*)(aRow + k0 + 40 + 16 * h);
    b1.h[0] = *(const us8*)(bRow + k0 + 32 + 16 * h);
    b1.h[1] = *(const us8*)(bRow + k0 + 40 + 16 * h);
    acc0 = __builtin_amdgcn_wmma_f32_16x16x32_bf16(false, a0.v, false, b0.v,
                                                   (short)0, acc0, false, false);
    acc1 = __builtin_amdgcn_wmma_f32_16x16x32_bf16(false, a1.v, false, b1.v,
                                                   (short)0, acc1, false, false);
  }
}

// ---------------------------------------------------------------------------
// Workspace layout (bytes)
// ---------------------------------------------------------------------------
#define RT_OFF   (0)                               // Rt  bf16 [RES][RES]     8 MiB
#define WT_OFF   (RT_OFF + RES * RES * 2)          // Wt  bf16 [RES][EMBED]   2 MiB
#define S0_OFF   (WT_OFF + RES * EMBED * 2)        // state buf0 bf16 [B][RES]
#define S1_OFF   (S0_OFF + BATCH * RES * 2)        // state buf1 bf16 [B][RES]
#define SF_OFF   (S1_OFF + BATCH * RES * 2)        // final state f32 [B][RES]
#define CTR_OFF  (SF_OFF + BATCH * RES * 4)        // 4 barrier domains, padded
#define EB_OFF   (CTR_OFF + 1024)                  // ebf bf16 [SEQ][B][EMBED] 32 MiB

// ---------------------------------------------------------------------------
// Prep: transpose f32 [rows][cols] -> bf16 [cols][rows] via LDS tile
// ---------------------------------------------------------------------------
__global__ void transpose_f32_to_bf16(const float* __restrict__ src,
                                      unsigned short* __restrict__ dst,
                                      int rows, int cols) {
  __shared__ float tile[32][33];
  const int tilesX = cols >> 5;
  const int bx = blockIdx.x % tilesX;
  const int by = blockIdx.x / tilesX;
  const int tx = threadIdx.x & 31;
  const int ty = threadIdx.x >> 5;
  const int c0 = bx << 5, r0 = by << 5;
#pragma unroll
  for (int i = 0; i < 4; ++i)
    tile[ty + 8 * i][tx] = src[(size_t)(r0 + ty + 8 * i) * cols + (c0 + tx)];
  __syncthreads();
#pragma unroll
  for (int i = 0; i < 4; ++i) {
    const int r = ty + 8 * i;
    dst[(size_t)(c0 + r) * rows + (r0 + tx)] = f2bf(tile[tx][r]);
  }
}

// ---------------------------------------------------------------------------
// Prep: gather + convert token embeddings once: ebf[t][b][e] = bf16(embW[x[b,t]][e])
// ---------------------------------------------------------------------------
__global__ void gather_embed_bf16(const int* __restrict__ x,
                                  const float* __restrict__ embW,
                                  unsigned short* __restrict__ ebf) {
  const size_t idx = (size_t)blockIdx.x * 256 + threadIdx.x;  // < SEQ*BATCH*EMBED
  const int    e   = (int)(idx & (EMBED - 1));
  const size_t tb  = idx >> 9;                                // EMBED = 512
  const int    b   = (int)(tb & (BATCH - 1));                 // BATCH = 64
  const int    t   = (int)(tb >> 6);
  const int    tok = x[(size_t)b * SEQ + t];
  ebf[idx] = f2bf(embW[(size_t)tok * EMBED + e]);
}

// ---------------------------------------------------------------------------
// Prep: zero state buffers + barrier counters
// ---------------------------------------------------------------------------
__global__ void init_state(unsigned short* __restrict__ s0,
                           unsigned short* __restrict__ s1,
                           unsigned int* __restrict__ ctrs) {
  const int i = blockIdx.x * blockDim.x + threadIdx.x;
  if (i < BATCH * RES) { s0[i] = 0; s1[i] = 0; }
  if (i < 256) ctrs[i] = 0;
}

// ---------------------------------------------------------------------------
// Persistent reservoir scan.
// Grid: 64 WGs x 256 threads = 512 wave32, one wave per 16x16 output tile.
// The batch rows are independent, so the grid decomposes into 4 independent
// barrier domains of 16 WGs (one per 16-row M-group). Per step each wave:
//   acc  = u_t                      (carried in from the previous iteration)
//   wait (group barrier, gen >= t)  -> state[t-1] visible
//   acc += state @ R                (K=2048, 64 WMMAs, 2 chains)
//   state[t] = tanh(acc); store; arrive (split barrier: signal only)
//   u_{t+1} = emb_{t+1} @ Win       (16 WMMAs, overlaps barrier propagation)
// ---------------------------------------------------------------------------
__launch_bounds__(256, 1)
__global__ void reservoir_persistent(const unsigned short* __restrict__ ebf,
                                     const unsigned short* __restrict__ Rt,
                                     const unsigned short* __restrict__ Wt,
                                     unsigned short* __restrict__ sA,
                                     unsigned short* __restrict__ sB,
                                     float* __restrict__ sF,
                                     unsigned int* __restrict__ bar) {
  const int lane  = threadIdx.x & 31;
  const int wid   = blockIdx.x * 8 + (threadIdx.x >> 5);  // 0..511
  const int mi    = wid >> 7;                             // 0..3   (batch group)
  const int ni    = wid & 127;                            // 0..127 (N tile)
  const int mbase = mi << 4;
  const int nbase = ni << 4;
  const int m     = lane & 15;
  const int h     = lane >> 4;
  const int arow  = mbase + m;

  const unsigned short* browR = Rt + (size_t)(nbase + m) * RES;
  const unsigned short* browW = Wt + (size_t)(nbase + m) * EMBED;

  // per-group split barrier: 16 WGs per domain, 64B-padded counters
  unsigned int* cnt = bar + (blockIdx.x >> 4) * 32;
  unsigned int* gen = cnt + 16;

  const v8f vzero = {};

  // u_0 = emb_0 @ Win (state-independent head start)
  v8f accU0 = vzero, accU1 = vzero;
  panel_accum(accU0, accU1, ebf + (size_t)arow * EMBED, browW, EMBED, h);

  for (int t = 0; t < SEQ; ++t) {
    const unsigned short* sCur = (t & 1) ? sB : sA;
    unsigned short*       sNxt = (t & 1) ? sA : sB;

    v8f acc0 = accU0, acc1 = accU1;

    // ---- wait: state[t-1] of this batch group complete ----
    if (t > 0) {
      if (threadIdx.x == 0)
        while ((int)__atomic_load_n(gen, __ATOMIC_ACQUIRE) < t)
          __builtin_amdgcn_s_sleep(1);
      __syncthreads();
      __threadfence();
    }

    // ---- state @ R over K = 2048 ----
    panel_accum(acc0, acc1, sCur + (size_t)arow * RES, browR, RES, h);

    // ---- tanh + scatter (C/D layout: elem j -> row mbase+8h+j, col nbase+m) ----
    unsigned short* outp  = sNxt + (size_t)(mbase + 8 * h) * RES + (nbase + m);
    float*          foutp = sF   + (size_t)(mbase + 8 * h) * RES + (nbase + m);
#pragma unroll
    for (int j = 0; j < 8; ++j) {
      const float v = fast_tanh(acc0[j] + acc1[j]);
      outp[(size_t)j * RES] = f2bf(v);
      if (t == SEQ - 1) foutp[(size_t)j * RES] = v;  // full-precision final state
    }

    // ---- arrive (signal step t done; do NOT wait here) ----
    __threadfence();
    __syncthreads();
    if (threadIdx.x == 0) {
      const unsigned int arrived = __atomic_fetch_add(cnt, 1u, __ATOMIC_ACQ_REL) + 1u;
      if (arrived == 16u) {
        __atomic_store_n(cnt, 0u, __ATOMIC_RELAXED);
        __atomic_fetch_add(gen, 1u, __ATOMIC_RELEASE);  // gen = t+1
      }
    }

    // ---- overlap barrier propagation with u_{t+1} projection ----
    if (t + 1 < SEQ) {
      accU0 = vzero; accU1 = vzero;
      panel_accum(accU0, accU1,
                  ebf + ((size_t)(t + 1) * BATCH + arow) * EMBED, browW, EMBED, h);
    }
  }
}

// ---------------------------------------------------------------------------
// Final LayerNorm over last dim of the f32 state [BATCH][RES]
// ---------------------------------------------------------------------------
__global__ void layernorm_out(const float* __restrict__ sF,
                              const float* __restrict__ gamma,
                              const float* __restrict__ beta,
                              float* __restrict__ out) {
  __shared__ float sh1[256];
  __shared__ float sh2[256];
  const int b = blockIdx.x;
  const float* row = sF + (size_t)b * RES;
  float s = 0.f, ss = 0.f;
  for (int i = threadIdx.x; i < RES; i += 256) {
    const float v = row[i];
    s += v; ss += v * v;
  }
  sh1[threadIdx.x] = s;
  sh2[threadIdx.x] = ss;
  __syncthreads();
  for (int off = 128; off > 0; off >>= 1) {
    if ((int)threadIdx.x < off) {
      sh1[threadIdx.x] += sh1[threadIdx.x + off];
      sh2[threadIdx.x] += sh2[threadIdx.x + off];
    }
    __syncthreads();
  }
  const float mu   = sh1[0] * (1.0f / RES);
  const float var  = sh2[0] * (1.0f / RES) - mu * mu;
  const float rstd = rsqrtf(var + LN_EPS);
  for (int i = threadIdx.x; i < RES; i += 256)
    out[(size_t)b * RES + i] = (row[i] - mu) * rstd * gamma[i] + beta[i];
}

// ---------------------------------------------------------------------------
// Launch
// ---------------------------------------------------------------------------
extern "C" void kernel_launch(void* const* d_in, const int* in_sizes, int n_in,
                              void* d_out, int out_size, void* d_ws, size_t ws_size,
                              hipStream_t stream) {
  const int*   x     = (const int*)d_in[0];
  const float* embW  = (const float*)d_in[1];
  const float* R     = (const float*)d_in[2];
  const float* Win   = (const float*)d_in[3];
  const float* gamma = (const float*)d_in[4];
  const float* beta  = (const float*)d_in[5];

  char* ws = (char*)d_ws;
  unsigned short* Rt  = (unsigned short*)(ws + RT_OFF);
  unsigned short* Wt  = (unsigned short*)(ws + WT_OFF);
  unsigned short* s0  = (unsigned short*)(ws + S0_OFF);
  unsigned short* s1  = (unsigned short*)(ws + S1_OFF);
  float*          sF  = (float*)(ws + SF_OFF);
  unsigned int*   ctr = (unsigned int*)(ws + CTR_OFF);
  unsigned short* ebf = (unsigned short*)(ws + EB_OFF);

  // prep: bf16 operand layouts (L2-resident), gathered embeddings, zero state
  transpose_f32_to_bf16<<<(RES / 32) * (RES / 32), 256, 0, stream>>>(R, Rt, RES, RES);
  transpose_f32_to_bf16<<<(EMBED / 32) * (RES / 32), 256, 0, stream>>>(Win, Wt, EMBED, RES);
  init_state<<<(BATCH * RES + 255) / 256, 256, 0, stream>>>(s0, s1, ctr);
  gather_embed_bf16<<<(SEQ * BATCH * EMBED) / 256, 256, 0, stream>>>(x, embW, ebf);

  // sequential scan: persistent kernel, 4 independent 16-WG barrier domains
  reservoir_persistent<<<64, 256, 0, stream>>>(ebf, Rt, Wt, s0, s1, sF, ctr);

  // final LayerNorm -> d_out (f32 [64][2048])
  layernorm_out<<<BATCH, 256, 0, stream>>>(sF, gamma, beta, (float*)d_out);
}
